// MIMICLSTMModel_41738492183164
// MI455X (gfx1250) — matrix operand order repaired and checked by reading, not compiled, not run on hardware
//
#include <hip/hip_runtime.h>
#include <hip/hip_bf16.h>

// ---------------------------------------------------------------------------
// Fused LSTM for MI455X (gfx1250):  B=4096, T=200, F=76, H=128 (4H=512), S=17
// One workgroup (256 thr = 8 waves, wave32) owns 16 batch rows for all T.
// ALL 28 weight B-fragments (W_hh: 16, W_ih: 12) are held in VGPRs for the
// whole 200-step recurrence (~224 VGPRs; gfx1250 has a 1024-VGPR file and the
// compiler addresses >v255 via s_set_vgpr_msb). Per-step LDS traffic is only
// the A fragments of h/x plus the h writeback. LSTM cell runs in registers.
// One barrier per timestep via double-buffered h and x.
// ---------------------------------------------------------------------------

typedef __attribute__((ext_vector_type(16))) __bf16 v16bf;
typedef __attribute__((ext_vector_type(8)))  __bf16 v8bf;
typedef __attribute__((ext_vector_type(8)))  float  v8f;

#define B_TOT   4096
#define T_STEPS 200
#define F_IN    76
#define FP      96      // F padded to multiple of 32 for WMMA K
#define H_DIM   128
#define G_DIM   512     // 4*H
#define S_DIM   17
#define MB      16      // batch rows per workgroup
#define NTHREADS 256

__device__ __forceinline__ __bf16 f2bf(float f) {
  unsigned u = __builtin_bit_cast(unsigned, f);
  u = (u + 0x7FFFu + ((u >> 16) & 1u)) >> 16;   // round-to-nearest-even
  unsigned short s = (unsigned short)u;
  return __builtin_bit_cast(__bf16, s);
}

__device__ __forceinline__ float fast_sig(float x) {
  return 1.0f / (1.0f + __expf(-x));
}
__device__ __forceinline__ float fast_tanh(float x) {
  x = fminf(15.0f, fmaxf(-15.0f, x));
  float e = __expf(2.0f * x);
  return (e - 1.0f) / (e + 1.0f);
}

// A fragment (16xK, bf16) from LDS, ISA 7.12.2 layout:
// lanes 0-15 : row M=lane,    halves {K=k0..k0+7,   K=k0+16..k0+23}
// lanes 16-31: row M=lane-16, halves {K=k0+8..k0+15, K=k0+24..k0+31}
__device__ __forceinline__ v16bf load_A(const __bf16* buf, int ldk, int lane, int k0) {
  int m  = lane & 15;
  int kb = k0 + ((lane >> 4) << 3);
  const __bf16* base = buf + m * ldk + kb;
  v8bf lo = *(const v8bf*)(base);
  v8bf hi = *(const v8bf*)(base + 16);
  v16bf r;
#pragma unroll
  for (int i = 0; i < 8; ++i) { r[i] = lo[i]; r[i + 8] = hi[i]; }
  return r;
}

// B fragment (Kx16, bf16) built from GLOBAL fp32 row-major W[N][kmax]:
// B(k,n) = W[n][k].  lanes 0-15: col N=n0+lane, K={k0..k0+7, k0+16..k0+23} ;
// lanes 16-31: col N=n0+lane-16, K={k0+8..k0+15, k0+24..k0+31}.  k >= kmax -> 0 (padding).
__device__ __forceinline__ v16bf load_B_global(const float* __restrict__ w, int kmax,
                                               int lane, int n0, int k0) {
  int n  = n0 + (lane & 15);
  int kb = k0 + ((lane >> 4) << 3);
  v16bf r;
#pragma unroll
  for (int i = 0; i < 16; ++i) {
    int k = kb + (i < 8 ? i : i + 8);
    float v = (k < kmax) ? w[(size_t)n * kmax + k] : 0.0f;
    r[i] = f2bf(v);
  }
  return r;
}

__global__ __launch_bounds__(NTHREADS, 1)
void lstm_fused(const float* __restrict__ x,      // [B,T,F]
                const float* __restrict__ stat,   // [B,S]
                const float* __restrict__ h0,     // [B,H]
                const float* __restrict__ c0,     // [B,H]
                const float* __restrict__ W_ih,   // [4H,F]
                const float* __restrict__ W_hh,   // [4H,H]
                const float* __restrict__ b_ih,   // [4H]
                const float* __restrict__ b_hh,   // [4H]
                const float* __restrict__ W_lin,  // [1,H+S]
                const float* __restrict__ b_lin,  // [1]
                float* __restrict__ out)          // [B,1]
{
  __shared__ __align__(16) __bf16 s_h[2][MB * H_DIM];     //  8 KB (double buffer)
  __shared__ __align__(16) __bf16 s_x[2][MB * FP];        //  6 KB (double buffer)
  __shared__ __align__(16) float  s_hf[MB * H_DIM];       //  8 KB (final fp32 h)
  __shared__ __align__(16) float  s_red[NTHREADS];        //  1 KB

  const int tid   = threadIdx.x;
  const int b0    = blockIdx.x * MB;
  const int wave  = tid >> 5;
  const int lane  = tid & 31;
  const int jcol  = wave * 16 + (lane & 15);  // hidden column this lane owns
  const int mbase = (lane >> 4) << 3;         // rows mbase..mbase+7 in acc VGPRs

  // ---- One-time setup ----
  // Loop-invariant weight fragments -> registers (held across all 200 steps).
  v16bf Bh[16];   // W_hh: [kt*4+p], kt=0..3 (K=128), p=gate i/f/g/o
#pragma unroll
  for (int kt = 0; kt < 4; ++kt)
#pragma unroll
    for (int p = 0; p < 4; ++p)
      Bh[kt * 4 + p] = load_B_global(W_hh, H_DIM, lane, p * H_DIM + wave * 16, kt * 32);

  v16bf Bx[12];   // W_ih: [kt*4+p], kt=0..2 (K=96 padded from 76)
#pragma unroll
  for (int kt = 0; kt < 3; ++kt)
#pragma unroll
    for (int p = 0; p < 4; ++p)
      Bx[kt * 4 + p] = load_B_global(W_ih, F_IN, lane, p * H_DIM + wave * 16, kt * 32);

  // per-lane persistent registers: combined bias (4) and cell state c (8)
  float bias_r[4];
#pragma unroll
  for (int p = 0; p < 4; ++p)
    bias_r[p] = b_ih[p * H_DIM + jcol] + b_hh[p * H_DIM + jcol];
  float c_reg[8];
#pragma unroll
  for (int v = 0; v < 8; ++v)
    c_reg[v] = c0[(size_t)(b0 + mbase + v) * H_DIM + jcol];

  // initial h and x padding
  for (int idx = tid; idx < MB * H_DIM; idx += NTHREADS)
    s_h[0][idx] = f2bf(h0[(size_t)b0 * H_DIM + idx]);    // rows contiguous
  for (int idx = tid; idx < 2 * MB * FP; idx += NTHREADS)
    ((__bf16*)s_x)[idx] = f2bf(0.0f);                    // zero incl. padding

  // precompute x-staging slots (MB*F_IN = 1216 elems over 256 threads)
  int    nslots = 0;
  size_t xoff[5];
  int    xdst[5];
#pragma unroll
  for (int k = 0; k < 5; ++k) {
    int idx = tid + k * NTHREADS;
    if (idx < MB * F_IN) {
      int r = idx / F_IN, f = idx - r * F_IN;
      xoff[nslots] = (size_t)(b0 + r) * T_STEPS * F_IN + f;
      xdst[nslots] = r * FP + f;
      ++nslots;
    }
  }
  __syncthreads();

  // stage x for t=0 into buffer 0
  for (int k = 0; k < nslots; ++k)
    s_x[0][xdst[k]] = f2bf(x[xoff[k]]);
  __syncthreads();

  // ---- Recurrence over T: ONE barrier per step ----
  for (int t = 0; t < T_STEPS; ++t) {
    const int sel  = t & 1;
    const int nsel = sel ^ 1;

    // gates[16, jcol + p*128] for p=0..3 accumulated in registers
    v8f acc[4];
#pragma unroll
    for (int p = 0; p < 4; ++p)
#pragma unroll
      for (int v = 0; v < 8; ++v) acc[p][v] = 0.0f;

#pragma unroll
    for (int kt = 0; kt < 4; ++kt) {                      // h @ W_hh^T (K=128)
      v16bf a = load_A(s_h[sel], H_DIM, lane, kt * 32);
#pragma unroll
      for (int p = 0; p < 4; ++p)
        acc[p] = __builtin_amdgcn_wmma_f32_16x16x32_bf16(
            false, a, false, Bh[kt * 4 + p], (short)0, acc[p], false, false);
    }
#pragma unroll
    for (int kt = 0; kt < 3; ++kt) {                      // x_t @ W_ih^T (K=96)
      v16bf a = load_A(s_x[sel], FP, lane, kt * 32);
#pragma unroll
      for (int p = 0; p < 4; ++p)
        acc[p] = __builtin_amdgcn_wmma_f32_16x16x32_bf16(
            false, a, false, Bx[kt * 4 + p], (short)0, acc[p], false, false);
    }

    // LSTM cell fully in registers: acc[0..3] = i,f,g,o for (row mbase+v, col jcol)
#pragma unroll
    for (int v = 0; v < 8; ++v) {
      float i = fast_sig (acc[0][v] + bias_r[0]);
      float f = fast_sig (acc[1][v] + bias_r[1]);
      float g = fast_tanh(acc[2][v] + bias_r[2]);
      float o = fast_sig (acc[3][v] + bias_r[3]);
      float c = f * c_reg[v] + i * g;
      float h = o * fast_tanh(c);
      c_reg[v] = c;
      s_h[nsel][(mbase + v) * H_DIM + jcol] = f2bf(h);    // for next step's A
      if (t == T_STEPS - 1) s_hf[(mbase + v) * H_DIM + jcol] = h;
    }

    // stage x for t+1 into the other buffer
    if (t + 1 < T_STEPS) {
      for (int k = 0; k < nslots; ++k)
        s_x[nsel][xdst[k]] = f2bf(x[xoff[k] + (size_t)(t + 1) * F_IN]);
      if (tid < MB && t + 2 < T_STEPS)
        __builtin_prefetch(x + ((size_t)(b0 + tid) * T_STEPS + (t + 2)) * F_IN, 0, 1);
    }
    __syncthreads();
  }

  // ---- Output head: out[b] = h . W_lin[0:H] + static . W_lin[H:H+S] + b_lin ----
  {
    int r  = tid >> 4;
    int jb = (tid & 15) * 8;
    float part = 0.0f;
#pragma unroll
    for (int q = 0; q < 8; ++q)
      part += s_hf[r * H_DIM + jb + q] * W_lin[jb + q];
    s_red[tid] = part;
  }
  __syncthreads();
  if (tid < MB) {
    float acc = b_lin[0];
#pragma unroll
    for (int k = 0; k < 16; ++k) acc += s_red[tid * 16 + k];
    for (int s = 0; s < S_DIM; ++s)
      acc += stat[(size_t)(b0 + tid) * S_DIM + s] * W_lin[H_DIM + s];
    out[b0 + tid] = acc;
  }
}

extern "C" void kernel_launch(void* const* d_in, const int* in_sizes, int n_in,
                              void* d_out, int out_size, void* d_ws, size_t ws_size,
                              hipStream_t stream) {
  (void)in_sizes; (void)n_in; (void)out_size; (void)d_ws; (void)ws_size;
  const float* x     = (const float*)d_in[0];
  const float* stat  = (const float*)d_in[1];
  const float* h0    = (const float*)d_in[2];
  const float* c0    = (const float*)d_in[3];
  const float* W_ih  = (const float*)d_in[4];
  const float* W_hh  = (const float*)d_in[5];
  const float* b_ih  = (const float*)d_in[6];
  const float* b_hh  = (const float*)d_in[7];
  const float* W_lin = (const float*)d_in[8];
  const float* b_lin = (const float*)d_in[9];
  float* out = (float*)d_out;

  dim3 grid(B_TOT / MB);     // 256 workgroups
  dim3 block(NTHREADS);      // 8 waves (wave32)
  lstm_fused<<<grid, block, 0, stream>>>(x, stat, h0, c0, W_ih, W_hh,
                                         b_ih, b_hh, W_lin, b_lin, out);
}
